// GRU_RPM_84121229459621
// MI455X (gfx1250) — matrix operand order repaired and checked by reading, not compiled, not run on hardware
//
#include <hip/hip_runtime.h>
#include <hip/hip_bf16.h>
#include <math.h>

// ---------------------------------------------------------------------------
// Shapes (from reference)
// ---------------------------------------------------------------------------
#define BATCH      64
#define TSTEPS     1000
#define NROWS      (BATCH * TSTEPS)     // 64000
#define XDIM       128
#define HDIM       256
#define CDIM       256
#define ZDIM       16
#define OUTDIM     152                  // 16 + 136
#define OUTPAD     160                  // 10 x 16 col tiles (cols 152..159 zero)
#define TRI        136
#define LN_EPS     1e-6f

typedef __attribute__((ext_vector_type(16))) _Float16 v16h;
typedef __attribute__((ext_vector_type(8)))  _Float16 v8h;
typedef __attribute__((ext_vector_type(8)))  float    v8f;

// ---------------------------------------------------------------------------
// Kernel 1: GRU scan. One workgroup of 256 threads; thread j owns channel j.
// carry[t] depends only on carry0/weights (not batch, not x).
// ---------------------------------------------------------------------------
__global__ void gru_scan_kernel(const float* __restrict__ carry0,
                                const float* __restrict__ Wr, const float* __restrict__ br,
                                const float* __restrict__ lnrg, const float* __restrict__ lnrb,
                                const float* __restrict__ Wz, const float* __restrict__ bz,
                                const float* __restrict__ lnzg, const float* __restrict__ lnzb,
                                const float* __restrict__ Wn, const float* __restrict__ bn,
                                const float* __restrict__ lnng, const float* __restrict__ lnnb,
                                float* __restrict__ carry_out)
{
    __shared__ float h[CDIM];
    __shared__ float red[6][CDIM];

    const int j = threadIdx.x;
    float hj = carry0[j];
    h[j] = hj;
    __syncthreads();

    for (int t = 0; t < TSTEPS; ++t) {
        float ar = br[j], az = bz[j], an = bn[j];
        for (int i = 0; i < CDIM; ++i) {
            float hv = h[i];
            ar = fmaf(hv, Wr[i * CDIM + j], ar);
            az = fmaf(hv, Wz[i * CDIM + j], az);
            an = fmaf(hv, Wn[i * CDIM + j], an);
        }
        red[0][j] = ar; red[1][j] = ar * ar;
        red[2][j] = az; red[3][j] = az * az;
        red[4][j] = an; red[5][j] = an * an;
        __syncthreads();
        for (int s = CDIM / 2; s > 0; s >>= 1) {
            if (j < s) {
                #pragma unroll
                for (int q = 0; q < 6; ++q) red[q][j] += red[q][j + s];
            }
            __syncthreads();
        }
        const float inv_n = 1.0f / (float)CDIM;
        float mr = red[0][0] * inv_n, vr = red[1][0] * inv_n - mr * mr;
        float mz = red[2][0] * inv_n, vz = red[3][0] * inv_n - mz * mz;
        float mn = red[4][0] * inv_n, vn = red[5][0] * inv_n - mn * mn;

        float lr = (ar - mr) * rsqrtf(vr + LN_EPS) * lnrg[j] + lnrb[j];
        float lz = (az - mz) * rsqrtf(vz + LN_EPS) * lnzg[j] + lnzb[j];
        float ln = (an - mn) * rsqrtf(vn + LN_EPS) * lnng[j] + lnnb[j];

        float r = 1.0f / (1.0f + expf(-lr));
        float z = 1.0f / (1.0f + expf(-lz));
        float n = tanhf(r * ln);
        float hn = (1.0f - z) * n + z * hj;

        carry_out[t * CDIM + j] = hn;
        __syncthreads();
        h[j] = hn;
        hj = hn;
        __syncthreads();
    }
}

// ---------------------------------------------------------------------------
// Kernel 2: carry_proj(t,n) = carry[t] @ Wo_top  (1000 x 152). Tiny; VALU.
// ---------------------------------------------------------------------------
__global__ void carry_proj_kernel(const float* __restrict__ carry,
                                  const float* __restrict__ Wo,
                                  float* __restrict__ cp)
{
    int i = blockIdx.x * blockDim.x + threadIdx.x;
    if (i >= TSTEPS * OUTDIM) return;
    int t = i / OUTDIM, n = i - t * OUTDIM;
    float s = 0.0f;
    for (int k = 0; k < CDIM; ++k)
        s = fmaf(carry[t * CDIM + k], Wo[k * OUTDIM + n], s);
    cp[i] = s;
}

// ---------------------------------------------------------------------------
// Prep: pack W (K x Nsrc f32, row stride srcStride) -> WT (N x K f16, col-major),
// zero-filling columns n >= Nsrc. Unconditional load via index clamp.
// ---------------------------------------------------------------------------
__global__ void pack_wt_kernel(const float* __restrict__ W, _Float16* __restrict__ WT,
                               int K, int N, int Nsrc, int srcStride)
{
    int i = blockIdx.x * blockDim.x + threadIdx.x;
    if (i >= K * N) return;
    int n = i / K, k = i - n * K;
    int nc = (n < Nsrc) ? n : (Nsrc - 1);
    float v = W[(size_t)k * srcStride + nc];
    WT[(size_t)n * K + k] = (_Float16)((n < Nsrc) ? v : 0.0f);
}

// Prep: f32 -> f16 elementwise (x conversion)
__global__ void cvt_f16_kernel(const float* __restrict__ src, _Float16* __restrict__ dst,
                               int count)
{
    int i = blockIdx.x * blockDim.x + threadIdx.x;
    if (i < count) dst[i] = (_Float16)src[i];
}

// ---------------------------------------------------------------------------
// WMMA fragment layouts as used here (v_wmma_f32_16x16x32_f16, wave32; A/B share k order):
// A 16x32 (MxK): row = lane&15; elems 0..15 -> K = kb+16*hi+0..15
//   => two contiguous 8-half (16B) runs of a row-major f16 activation row.
// B 32x16 (KxN): col = lane&15; elems 0..15 -> K = kb+16*hi+0..15
//   => one contiguous 16-half (32B) run of a column-major f16 weight column.
// C/D 16x16:     elem r -> M = r + 8*hi, N = lane&15
// ---------------------------------------------------------------------------

// Kernel 3/4: GEMM (16 rows/block, 16 waves = 256 cols) + bias + row-LN + ReLU
template <int K>
__global__ __launch_bounds__(512)
void gemm_ln_relu_f16(const _Float16* __restrict__ Ah,  // (NROWS, K) f16 row-major
                      const _Float16* __restrict__ WT,  // (256, K) f16 col-major
                      const float* __restrict__ bias,
                      const float* __restrict__ g,
                      const float* __restrict__ bb,
                      _Float16* __restrict__ outh)      // (NROWS, 256) f16
{
    __shared__ __align__(16) _Float16 Alds[16][K + 8];
    __shared__ float Olds[16][HDIM + 1];

    const int tid  = threadIdx.x;
    const int wave = tid >> 5;           // 0..15 -> 16-col tile
    const int lane = tid & 31;
    const int rowbase = blockIdx.x * 16;

    // stage A tile (f16) into LDS, 8 halves (16B) per thread-iteration
    for (int i = tid; i < 16 * K / 8; i += blockDim.x) {
        int r = (i * 8) / K, c = (i * 8) % K;
        *(v8h*)&Alds[r][c] = *(const v8h*)(Ah + (size_t)(rowbase + r) * K + c);
    }
    __syncthreads();

    const int mrow = lane & 15;
    const int hi   = lane >> 4;
    const int n    = wave * 16 + mrow;
    const _Float16* wcol = WT + (size_t)n * K;

    v8f acc = {0.f, 0.f, 0.f, 0.f, 0.f, 0.f, 0.f, 0.f};
    #pragma unroll
    for (int kb = 0; kb < K; kb += 32) {
        v8h a0 = *(const v8h*)&Alds[mrow][kb + 16 * hi];
        v8h a1 = *(const v8h*)&Alds[mrow][kb + 16 * hi + 8];
        v8h b0 = *(const v8h*)(wcol + kb + 16 * hi);
        v8h b1 = *(const v8h*)(wcol + kb + 16 * hi + 8);
        v16h a, b;
        #pragma unroll
        for (int e = 0; e < 8; ++e) {
            a[e] = a0[e]; a[e + 8] = a1[e];
            b[e] = b0[e]; b[e + 8] = b1[e];
        }
        acc = __builtin_amdgcn_wmma_f32_16x16x32_f16(false, a, false, b,
                                                     (short)0, acc, false, false);
    }

    #pragma unroll
    for (int r = 0; r < 8; ++r) {
        int m = r + (hi << 3);
        Olds[m][n] = acc[r] + bias[n];
    }
    __syncthreads();

    // LayerNorm over the 256-wide row; wave w owns row w. Output f16.
    {
        const int m = wave;
        float s = 0.f, ss = 0.f;
        for (int c = lane; c < HDIM; c += 32) {
            float v = Olds[m][c];
            s += v; ss += v * v;
        }
        #pragma unroll
        for (int off = 16; off > 0; off >>= 1) {
            s  += __shfl_xor(s,  off, 32);
            ss += __shfl_xor(ss, off, 32);
        }
        float mean = s * (1.0f / HDIM);
        float var  = ss * (1.0f / HDIM) - mean * mean;
        float inv  = rsqrtf(var + LN_EPS);
        for (int c = lane; c < HDIM; c += 32) {
            float v = (Olds[m][c] - mean) * inv * g[c] + bb[c];
            outh[(size_t)(rowbase + m) * HDIM + c] = (_Float16)fmaxf(v, 0.0f);
        }
    }
}

// ---------------------------------------------------------------------------
// Kernel 5: out = hx1 @ Wo_bot + carry_proj[t] + bo   (64000 x 152)
// Wo_bot pre-packed f16 col-major, padded to 160 cols (152..159 are zero),
// so the mainloop has no masking; only the final store is masked.
// ---------------------------------------------------------------------------
__global__ __launch_bounds__(320)
void gemm_out_f16(const _Float16* __restrict__ Ah,   // (NROWS, 256) f16
                  const _Float16* __restrict__ WoT,  // (160, 256) f16 col-major
                  const float* __restrict__ bo,      // (152)
                  const float* __restrict__ cproj,   // (1000, 152)
                  float* __restrict__ out_lin)       // (NROWS, 152) f32
{
    __shared__ __align__(16) _Float16 Alds[16][HDIM + 8];

    const int tid  = threadIdx.x;
    const int wave = tid >> 5;           // 0..9
    const int lane = tid & 31;
    const int rowbase = blockIdx.x * 16;

    for (int i = tid; i < 16 * HDIM / 8; i += blockDim.x) {
        int r = (i * 8) / HDIM, c = (i * 8) % HDIM;
        *(v8h*)&Alds[r][c] = *(const v8h*)(Ah + (size_t)(rowbase + r) * HDIM + c);
    }
    __syncthreads();

    const int mrow = lane & 15;
    const int hi   = lane >> 4;
    const int n    = wave * 16 + mrow;   // 0..159
    const _Float16* wcol = WoT + (size_t)n * HDIM;

    v8f acc = {0.f, 0.f, 0.f, 0.f, 0.f, 0.f, 0.f, 0.f};
    #pragma unroll
    for (int kb = 0; kb < HDIM; kb += 32) {
        v8h a0 = *(const v8h*)&Alds[mrow][kb + 16 * hi];
        v8h a1 = *(const v8h*)&Alds[mrow][kb + 16 * hi + 8];
        v8h b0 = *(const v8h*)(wcol + kb + 16 * hi);
        v8h b1 = *(const v8h*)(wcol + kb + 16 * hi + 8);
        v16h a, b;
        #pragma unroll
        for (int e = 0; e < 8; ++e) {
            a[e] = a0[e]; a[e + 8] = a1[e];
            b[e] = b0[e]; b[e + 8] = b1[e];
        }
        acc = __builtin_amdgcn_wmma_f32_16x16x32_f16(false, a, false, b,
                                                     (short)0, acc, false, false);
    }

    if (n < OUTDIM) {
        #pragma unroll
        for (int r = 0; r < 8; ++r) {
            int m   = r + (hi << 3);
            int row = rowbase + m;
            int t   = row % TSTEPS;
            out_lin[(size_t)row * OUTDIM + n] = acc[r] + bo[n] + cproj[t * OUTDIM + n];
        }
    }
}

// ---------------------------------------------------------------------------
// Kernel 6: Cholesky head. 16 lanes per (b,t) item; lane = column index.
// ---------------------------------------------------------------------------
#define HGROUPS 16
__global__ __launch_bounds__(256)
void head_kernel(const float* __restrict__ out_lin,
                 float* __restrict__ sigma_o,   // (NROWS,16,16)
                 float* __restrict__ mu_o,      // (NROWS,16)
                 float* __restrict__ j_o,       // (NROWS,16,16)
                 float* __restrict__ h_o)       // (NROWS,16)
{
    __shared__ float Lm[HGROUPS][ZDIM][ZDIM + 1];
    __shared__ float Li[HGROUPS][ZDIM][ZDIM + 1];
    __shared__ float Jm[HGROUPS][ZDIM][ZDIM + 1];
    __shared__ float Mu[HGROUPS][ZDIM];

    const int g = threadIdx.x >> 4;
    const int l = threadIdx.x & 15;
    const size_t item = (size_t)blockIdx.x * HGROUPS + g;
    const float* row = out_lin + item * OUTDIM;

    float muv = row[l];
    Mu[g][l] = muv;
    mu_o[item * ZDIM + l] = muv;

    #pragma unroll
    for (int k = 0; k < ZDIM; ++k) Lm[g][l][k] = 0.0f;
    __syncthreads();

    // scatter tril(flat) -> L  (linear k over (r,c) pairs, c<=r, row-major)
    for (int k = l; k < TRI; k += 16) {
        int r = 0;
        while ((r + 1) * (r + 2) / 2 <= k) ++r;
        int c = k - r * (r + 1) / 2;
        Lm[g][r][c] = row[ZDIM + k];
    }
    __syncthreads();

    // softplus on diagonal
    {
        float d  = Lm[g][l][l];
        float sp = fmaxf(d, 0.0f) + log1pf(expf(-fabsf(d)));
        Lm[g][l][l] = sp;
    }
    __syncthreads();

    float Lr[ZDIM];
    #pragma unroll
    for (int k = 0; k < ZDIM; ++k) Lr[k] = Lm[g][l][k];

    // Sigma[i][l] = sum_k L[i][k] * L[l][k]
    #pragma unroll
    for (int i = 0; i < ZDIM; ++i) {
        float s = 0.0f;
        #pragma unroll
        for (int k = 0; k < ZDIM; ++k) s = fmaf(Lm[g][i][k], Lr[k], s);
        sigma_o[item * (ZDIM * ZDIM) + i * ZDIM + l] = s;
    }

    // Forward substitution: column l of Linv (solve L y = e_l)
    float y[ZDIM];
    #pragma unroll
    for (int i = 0; i < ZDIM; ++i) {
        float s = (i == l) ? 1.0f : 0.0f;
        #pragma unroll
        for (int k = 0; k < ZDIM; ++k)
            if (k < i) s = fmaf(-Lm[g][i][k], y[k], s);
        y[i] = s / Lm[g][i][i];
    }
    #pragma unroll
    for (int k = 0; k < ZDIM; ++k) Li[g][k][l] = y[k];
    __syncthreads();

    // J[i][l] = sum_k Linv[k][i] * Linv[k][l]
    #pragma unroll
    for (int i = 0; i < ZDIM; ++i) {
        float s = 0.0f;
        #pragma unroll
        for (int k = 0; k < ZDIM; ++k) s = fmaf(Li[g][k][i], y[k], s);
        j_o[item * (ZDIM * ZDIM) + i * ZDIM + l] = s;
        Jm[g][i][l] = s;
    }
    __syncthreads();

    // h_nat[l] = sum_j J[l][j] * mu[j]
    {
        float s = 0.0f;
        #pragma unroll
        for (int jx = 0; jx < ZDIM; ++jx) s = fmaf(Jm[g][l][jx], Mu[g][jx], s);
        h_o[item * ZDIM + l] = s;
    }
}

// ---------------------------------------------------------------------------
// Launch
// ---------------------------------------------------------------------------
extern "C" void kernel_launch(void* const* d_in, const int* in_sizes, int n_in,
                              void* d_out, int out_size, void* d_ws, size_t ws_size,
                              hipStream_t stream)
{
    const float* x      = (const float*)d_in[0];
    const float* carry0 = (const float*)d_in[1];
    const float* Wr   = (const float*)d_in[2];
    const float* br   = (const float*)d_in[3];
    const float* lnrg = (const float*)d_in[4];
    const float* lnrb = (const float*)d_in[5];
    const float* Wz   = (const float*)d_in[6];
    const float* bz   = (const float*)d_in[7];
    const float* lnzg = (const float*)d_in[8];
    const float* lnzb = (const float*)d_in[9];
    const float* Wn   = (const float*)d_in[10];
    const float* bn   = (const float*)d_in[11];
    const float* lnng = (const float*)d_in[12];
    const float* lnnb = (const float*)d_in[13];
    const float* W0   = (const float*)d_in[14];
    const float* b0   = (const float*)d_in[15];
    const float* ln0g = (const float*)d_in[16];
    const float* ln0b = (const float*)d_in[17];
    const float* W1   = (const float*)d_in[18];
    const float* b1   = (const float*)d_in[19];
    const float* ln1g = (const float*)d_in[20];
    const float* ln1b = (const float*)d_in[21];
    const float* Wo   = (const float*)d_in[22];
    const float* bo   = (const float*)d_in[23];

    // workspace layout (bytes)
    char* wsb = (char*)d_ws;
    size_t off = 0;
    float* carry = (float*)(wsb + off);            off += (size_t)TSTEPS * CDIM * 4;
    float* cproj = (float*)(wsb + off);            off += (size_t)TSTEPS * OUTDIM * 4;
    _Float16* W0T = (_Float16*)(wsb + off);        off += (size_t)HDIM * XDIM * 2;
    _Float16* W1T = (_Float16*)(wsb + off);        off += (size_t)HDIM * HDIM * 2;
    _Float16* WoT = (_Float16*)(wsb + off);        off += (size_t)OUTPAD * HDIM * 2;
    _Float16* xh  = (_Float16*)(wsb + off);        off += (size_t)NROWS * XDIM * 2;
    _Float16* hx0 = (_Float16*)(wsb + off);        off += (size_t)NROWS * HDIM * 2;
    _Float16* hx1 = (_Float16*)(wsb + off);        off += (size_t)NROWS * HDIM * 2;
    float* out_lin = (float*)(wsb + off);          off += (size_t)NROWS * OUTDIM * 4;

    float* out   = (float*)d_out;
    float* sigma = out;
    float* mu    = sigma + (size_t)NROWS * ZDIM * ZDIM;
    float* Jo    = mu    + (size_t)NROWS * ZDIM;
    float* hnat  = Jo    + (size_t)NROWS * ZDIM * ZDIM;

    // 1) serial GRU scan (single workgroup)
    gru_scan_kernel<<<1, CDIM, 0, stream>>>(carry0, Wr, br, lnrg, lnrb,
                                            Wz, bz, lnzg, lnzb,
                                            Wn, bn, lnng, lnnb, carry);
    // 2) carry projection through Wo_top (rows 0..255 of Wo)
    {
        int n = TSTEPS * OUTDIM;
        carry_proj_kernel<<<(n + 255) / 256, 256, 0, stream>>>(carry, Wo, cproj);
    }
    // 3) pack weights to f16 col-major (zero-padded for Wo_bot)
    {
        int n0 = XDIM * HDIM;
        pack_wt_kernel<<<(n0 + 255) / 256, 256, 0, stream>>>(W0, W0T, XDIM, HDIM, HDIM, HDIM);
        int n1 = HDIM * HDIM;
        pack_wt_kernel<<<(n1 + 255) / 256, 256, 0, stream>>>(W1, W1T, HDIM, HDIM, HDIM, HDIM);
        int n2 = HDIM * OUTPAD;
        pack_wt_kernel<<<(n2 + 255) / 256, 256, 0, stream>>>(Wo + (size_t)CDIM * OUTDIM, WoT,
                                                             HDIM, OUTPAD, OUTDIM, OUTDIM);
        int n3 = NROWS * XDIM;
        cvt_f16_kernel<<<(n3 + 255) / 256, 256, 0, stream>>>(x, xh, n3);
    }
    // 4) MLP layer 0 (WMMA): hx0 = relu(ln(x @ W0 + b0))
    gemm_ln_relu_f16<XDIM><<<NROWS / 16, 512, 0, stream>>>(xh, W0T, b0, ln0g, ln0b, hx0);
    // 5) MLP layer 1 (WMMA): hx1 = relu(ln(hx0 @ W1 + b1))
    gemm_ln_relu_f16<HDIM><<<NROWS / 16, 512, 0, stream>>>(hx0, W1T, b1, ln1g, ln1b, hx1);
    // 6) output projection (WMMA) + carry part + bias
    gemm_out_f16<<<NROWS / 16, 320, 0, stream>>>(hx1, WoT, bo, cproj, out_lin);
    // 7) Cholesky head -> Sigma, mu, J, h_nat
    head_kernel<<<NROWS / HGROUPS, 256, 0, stream>>>(out_lin, sigma, mu, Jo, hnat);
}
